// VitonAttnProcessor_88742614270275
// MI455X (gfx1250) — matrix operand order repaired and checked by reading, not compiled, not run on hardware
//
#include <hip/hip_runtime.h>

// ---------------------------------------------------------------------------
// VitonAttnProcessor for MI455X (gfx1250, wave32, WMMA bf16 16x16x32)
// B=2, S2=2048, C=1280, NH=8, d=160
// ---------------------------------------------------------------------------

typedef __bf16 bf16_t;
typedef __attribute__((ext_vector_type(16))) __bf16 v16bf;
typedef __attribute__((ext_vector_type(8)))  float  v8f;

#define BATCH   2
#define SEQ     1024
#define SEQ2    2048
#define TOK     4096        // BATCH * SEQ2
#define CH      1280
#define NHEADS  8
#define HDIM    160
#define QSCALE  0.07905694150420949f   // 1/sqrt(160)

static __device__ __forceinline__ unsigned short f2bf(float f) {
  bf16_t b = (bf16_t)f;
  return __builtin_bit_cast(unsigned short, b);
}

union FragU { v16bf v; uint4 q[2]; };

// two 16B loads -> one 32B (16 x bf16) fragment
static __device__ __forceinline__ v16bf load_frag(const bf16_t* p0, const bf16_t* p1) {
  FragU f;
  f.q[0] = *reinterpret_cast<const uint4*>(p0);
  f.q[1] = *reinterpret_cast<const uint4*>(p1);
  return f.v;
}

// A fragment (16x32, row-major source, stride ld halves):
// lanes 0-15 : K = k0+0..7  and k0+16..23   (M = lane)
// lanes 16-31: K = k0+8..15 and k0+24..31   (M = lane-16)
static __device__ __forceinline__ v16bf load_a(const bf16_t* base, int ld,
                                               int m0, int k0, int lane) {
  int m  = m0 + (lane & 15);
  int ko = k0 + ((lane < 16) ? 0 : 8);
  const bf16_t* p = base + (size_t)m * ld + ko;
  return load_frag(p, p + 16);
}

// B fragment (32x16) from [N][K] storage (stride ld halves):
// lanes 0-15 : K = k0+0..7 and k0+16..23 (N = lane); lanes 16-31: K = k0+8..15 and k0+24..31
static __device__ __forceinline__ v16bf load_b(const bf16_t* base, int ld,
                                               int n0, int k0, int lane) {
  int n  = n0 + (lane & 15);
  int ko = k0 + ((lane < 16) ? 0 : 8);
  const bf16_t* p = base + (size_t)n * ld + ko;
  return load_frag(p, p + 16);
}

static __device__ __forceinline__ v8f wmma_bf16(v16bf a, v16bf b, v8f c) {
  return __builtin_amdgcn_wmma_f32_16x16x32_bf16(
      /*neg_a=*/false, a, /*neg_b=*/false, b,
      /*c_mod=*/(short)0, c, /*reuse_a=*/false, /*reuse_b=*/false);
}

static __device__ __forceinline__ uint4 pack8(const float* o) {
  uint4 r;
  r.x = (unsigned)f2bf(o[0]) | ((unsigned)f2bf(o[1]) << 16);
  r.y = (unsigned)f2bf(o[2]) | ((unsigned)f2bf(o[3]) << 16);
  r.z = (unsigned)f2bf(o[4]) | ((unsigned)f2bf(o[5]) << 16);
  r.w = (unsigned)f2bf(o[6]) | ((unsigned)f2bf(o[7]) << 16);
  return r;
}

// ---------------------------------------------------------------------------
// K1: f32 -> bf16 conversion of hidden_states (8 elems/thread, b128 stores)
// ---------------------------------------------------------------------------
__global__ void __launch_bounds__(256) k_cvt(const float* __restrict__ hs,
                                             bf16_t* __restrict__ hsb) {
  size_t i = (size_t)blockIdx.x * blockDim.x + threadIdx.x;  // group of 8
  size_t base = i * 8;
  float o[8];
  #pragma unroll
  for (int j = 0; j < 8; ++j) o[j] = hs[base + j];
  *reinterpret_cast<uint4*>(hsb + base) = pack8(o);
}

// ---------------------------------------------------------------------------
// K2: weight transpose + f32->bf16.  wT[z][n][k] = w_z[k][n]
// slots: 0 attn_wq, 1 proc_wq, 2 attn_wk, 3 proc_wk, 4 attn_wv, 5 proc_wv, 6 out_w
// ---------------------------------------------------------------------------
__global__ void __launch_bounds__(256) k_trans(
    const float* __restrict__ s0, const float* __restrict__ s1,
    const float* __restrict__ s2, const float* __restrict__ s3,
    const float* __restrict__ s4, const float* __restrict__ s5,
    const float* __restrict__ s6, bf16_t* __restrict__ wT) {
  __shared__ float tile[32][33];
  const float* src;
  switch (blockIdx.z) {
    case 0: src = s0; break; case 1: src = s1; break;
    case 2: src = s2; break; case 3: src = s3; break;
    case 4: src = s4; break; case 5: src = s5; break;
    default: src = s6; break;
  }
  bf16_t* dst = wT + (size_t)blockIdx.z * CH * CH;
  int tx = threadIdx.x, ty = threadIdx.y;
  int c0 = blockIdx.x * 32, r0 = blockIdx.y * 32;
  #pragma unroll
  for (int i = 0; i < 4; ++i) {
    int r = r0 + ty + i * 8;
    tile[ty + i * 8][tx] = src[(size_t)r * CH + (c0 + tx)];
  }
  __syncthreads();
  #pragma unroll
  for (int i = 0; i < 4; ++i) {
    int n = c0 + ty + i * 8;   // output row = source column
    int k = r0 + tx;           // output col = source row
    dst[(size_t)n * CH + k] = (bf16_t)tile[tx][ty + i * 8];
  }
}

// ---------------------------------------------------------------------------
// K3: gate  w = sigmoid([hs, mask] @ ww + wb), one wave per token row
// ---------------------------------------------------------------------------
__global__ void __launch_bounds__(128) k_gate(const float* __restrict__ hs,
                                              const float* __restrict__ src_mask,
                                              const float* __restrict__ ww,
                                              const float* __restrict__ wb,
                                              float* __restrict__ gate) {
  int row  = (blockIdx.x * blockDim.x + threadIdx.x) >> 5;
  int lane = threadIdx.x & 31;
  const float* r = hs + (size_t)row * CH;
  float s = 0.0f;
  #pragma unroll 4
  for (int j = lane; j < CH; j += 32) s += r[j] * ww[j];
  #pragma unroll
  for (int m = 16; m >= 1; m >>= 1) s += __shfl_xor(s, m, 32);
  int b  = row >> 11;
  int t1 = row & 1023;                 // token index mod S (both halves share m)
  int iy = t1 >> 5, jx = t1 & 31;      // 32x32 grid of 8-strided samples
  float mval = src_mask[(size_t)b * 65536 + (size_t)(iy * 8) * 256 + jx * 8];
  float x = s + mval * ww[CH] + wb[0];
  float g = 1.0f / (1.0f + __expf(-x));
  if (lane == 0) gate[row] = g;
}

// ---------------------------------------------------------------------------
// K4: fused QKV projection, templated on JOB so the K-loop is branch-free.
// grid = (CH/64, TOK/64), 4 waves/block, each wave one 16x64 tile.
//  JOB 0: Q = (hs@attn_wq)*w + (hs@proc_wq)*(1-w), *1/sqrt(d), bf16 row-major
//  JOB 1: K = hs@{attn_wk|proc_wk per half}, bf16 row-major  (== B^T layout)
//  JOB 2: V, stored transposed [b][dim][token] (B^T layout for P@V)
// ---------------------------------------------------------------------------
template <int JOB>
__global__ void __launch_bounds__(128) k_qkv(const bf16_t* __restrict__ hsb,
                                             const bf16_t* __restrict__ wT,
                                             const float* __restrict__ gate,
                                             bf16_t* __restrict__ outQ,
                                             bf16_t* __restrict__ outK,
                                             bf16_t* __restrict__ outV) {
  const int lane = threadIdx.x & 31;
  const int w    = threadIdx.x >> 5;
  const int m0   = (blockIdx.y * 4 + w) * 16;
  const int n0   = blockIdx.x * 64;
  const bool firstHalf = ((m0 & (SEQ2 - 1)) < SEQ);
  const size_t WSZ = (size_t)CH * CH;

  const bf16_t* WA;
  const bf16_t* W2 = wT;
  if constexpr (JOB == 0)      { WA = wT + 0 * WSZ; W2 = wT + 1 * WSZ; }
  else if constexpr (JOB == 1) { WA = wT + (firstHalf ? 2 : 3) * WSZ; }
  else                         { WA = wT + (firstHalf ? 4 : 5) * WSZ; }

  v8f z = {0.f,0.f,0.f,0.f,0.f,0.f,0.f,0.f};
  v8f acc[4]  = {z, z, z, z};
  v8f acc2[4] = {z, z, z, z};

  #pragma unroll 2
  for (int kc = 0; kc < CH / 32; ++kc) {
    int k0 = kc * 32;
    v16bf a = load_a(hsb, CH, m0, k0, lane);
    __builtin_prefetch(hsb + (size_t)(m0 + (lane & 15)) * CH + k0 + 64, 0, 1);
    #pragma unroll
    for (int j = 0; j < 4; ++j) {
      v16bf bb = load_b(WA, CH, n0 + 16 * j, k0, lane);
      acc[j] = wmma_bf16(a, bb, acc[j]);
      if constexpr (JOB == 0) {
        v16bf bp = load_b(W2, CH, n0 + 16 * j, k0, lane);
        acc2[j] = wmma_bf16(a, bp, acc2[j]);
      }
    }
  }

  const int rb  = (lane < 16) ? 0 : 8;
  const int nlo = lane & 15;

  if constexpr (JOB == 0) {
    float gv[8];
    #pragma unroll
    for (int i = 0; i < 8; ++i) gv[i] = gate[m0 + rb + i];
    #pragma unroll
    for (int j = 0; j < 4; ++j) {
      int n = n0 + 16 * j + nlo;
      #pragma unroll
      for (int i = 0; i < 8; ++i) {
        float g = gv[i];
        float q = (acc[j][i] * g + acc2[j][i] * (1.0f - g)) * QSCALE;
        outQ[(size_t)(m0 + rb + i) * CH + n] = (bf16_t)q;
      }
    }
  } else if constexpr (JOB == 1) {
    #pragma unroll
    for (int j = 0; j < 4; ++j) {
      int n = n0 + 16 * j + nlo;
      #pragma unroll
      for (int i = 0; i < 8; ++i)
        outK[(size_t)(m0 + rb + i) * CH + n] = (bf16_t)acc[j][i];
    }
  } else {
    int b  = m0 >> 11;
    int t0 = (m0 & (SEQ2 - 1)) + rb;
    #pragma unroll
    for (int j = 0; j < 4; ++j) {
      int n = n0 + 16 * j + nlo;
      float o[8];
      #pragma unroll
      for (int i = 0; i < 8; ++i) o[i] = acc[j][i];
      *reinterpret_cast<uint4*>(outV + ((size_t)b * CH + n) * SEQ2 + t0) = pack8(o);
    }
  }
}

// ---------------------------------------------------------------------------
// K5: flash attention.  One wave per (batch, head, 16-query tile).
// 32-key blocks; scores via 2x5 WMMA; online softmax (shfl_xor row reductions);
// P f32(C-layout) -> bf16 A-layout via LDS; P@V via 10 WMMA into 80 f32 accums.
// ---------------------------------------------------------------------------
__global__ void __launch_bounds__(128) k_attn(const bf16_t* __restrict__ Qb,
                                              const bf16_t* __restrict__ Kb,
                                              const bf16_t* __restrict__ Vt,
                                              bf16_t* __restrict__ AOb) {
  __shared__ bf16_t lp[4][16][32];
  const int lane = threadIdx.x & 31;
  const int w    = threadIdx.x >> 5;
  const int job  = blockIdx.x * 4 + w;           // 0..2047
  const int qt   = job & 127;
  const int h    = (job >> 7) & 7;
  const int b    = job >> 10;
  const int m0   = b * SEQ2 + qt * 16;           // global query row base
  const int kd0  = h * HDIM;                     // head's dim offset
  const bf16_t* Vb = Vt + (size_t)b * CH * SEQ2;

  v16bf qf[5];
  #pragma unroll
  for (int c = 0; c < 5; ++c) qf[c] = load_a(Qb, CH, m0, kd0 + 32 * c, lane);

  v8f z = {0.f,0.f,0.f,0.f,0.f,0.f,0.f,0.f};
  v8f acc[10] = {z,z,z,z,z,z,z,z,z,z};
  float mrow[8], lrow[8];
  #pragma unroll
  for (int i = 0; i < 8; ++i) { mrow[i] = -1e30f; lrow[i] = 0.0f; }

  const int rb  = (lane < 16) ? 0 : 8;
  const int nlo = lane & 15;

  for (int kb = 0; kb < SEQ2; kb += 32) {
    // ---- scores: 16 queries x 32 keys, K = 160 ----
    v8f s0 = z, s1 = z;
    #pragma unroll
    for (int c = 0; c < 5; ++c) {
      v16bf b0 = load_b(Kb, CH, b * SEQ2 + kb,      kd0 + 32 * c, lane);
      v16bf b1 = load_b(Kb, CH, b * SEQ2 + kb + 16, kd0 + 32 * c, lane);
      s0 = wmma_bf16(qf[c], b0, s0);
      s1 = wmma_bf16(qf[c], b1, s1);
    }
    // ---- online softmax ----
    float p0[8], p1[8], corr[8];
    #pragma unroll
    for (int i = 0; i < 8; ++i) {
      float v = fmaxf(s0[i], s1[i]);
      #pragma unroll
      for (int msk = 8; msk >= 1; msk >>= 1) v = fmaxf(v, __shfl_xor(v, msk, 32));
      float mnew = fmaxf(mrow[i], v);
      corr[i] = __expf(mrow[i] - mnew);
      mrow[i] = mnew;
      p0[i] = __expf(s0[i] - mnew);
      p1[i] = __expf(s1[i] - mnew);
      float rs = p0[i] + p1[i];
      #pragma unroll
      for (int msk = 8; msk >= 1; msk >>= 1) rs += __shfl_xor(rs, msk, 32);
      lrow[i] = lrow[i] * corr[i] + rs;
    }
    // ---- C-layout -> A-layout transpose of P through LDS (bf16) ----
    #pragma unroll
    for (int i = 0; i < 8; ++i) {
      lp[w][rb + i][nlo]      = (bf16_t)p0[i];
      lp[w][rb + i][16 + nlo] = (bf16_t)p1[i];
    }
    asm volatile("s_wait_dscnt 0" ::: "memory");  // per-wave LDS RAW fence
    v16bf pf;
    {
      int m  = lane & 15;
      int ko = (lane < 16) ? 0 : 8;
      const bf16_t* p = &lp[w][m][ko];
      pf = load_frag(p, p + 16);
    }
    // ---- rescale accumulators, then P@V ----
    #pragma unroll
    for (int nt = 0; nt < 10; ++nt) {
      #pragma unroll
      for (int i = 0; i < 8; ++i) acc[nt][i] *= corr[i];
    }
    #pragma unroll
    for (int nt = 0; nt < 10; ++nt) {
      v16bf vf = load_b(Vb, SEQ2, kd0 + nt * 16, kb, lane);
      acc[nt] = wmma_bf16(pf, vf, acc[nt]);
    }
  }

  // ---- normalize and store (bf16 row-major, A-layout for the final GEMM) ----
  float inv[8];
  #pragma unroll
  for (int i = 0; i < 8; ++i) inv[i] = 1.0f / lrow[i];
  #pragma unroll
  for (int nt = 0; nt < 10; ++nt) {
    int n = kd0 + nt * 16 + nlo;
    #pragma unroll
    for (int i = 0; i < 8; ++i)
      AOb[(size_t)(m0 + rb + i) * CH + n] = (bf16_t)(acc[nt][i] * inv[i]);
  }
}

// ---------------------------------------------------------------------------
// K6: out = attn_out @ out_w + out_b + hidden_states   (f32 output)
// ---------------------------------------------------------------------------
__global__ void __launch_bounds__(128) k_out(const bf16_t* __restrict__ AOb,
                                             const bf16_t* __restrict__ wTo,
                                             const float* __restrict__ out_b,
                                             const float* __restrict__ hs,
                                             float* __restrict__ out) {
  const int lane = threadIdx.x & 31;
  const int w    = threadIdx.x >> 5;
  const int m0   = (blockIdx.y * 4 + w) * 16;
  const int n0   = blockIdx.x * 64;

  v8f z = {0.f,0.f,0.f,0.f,0.f,0.f,0.f,0.f};
  v8f acc[4] = {z, z, z, z};
  #pragma unroll 2
  for (int kc = 0; kc < CH / 32; ++kc) {
    int k0 = kc * 32;
    v16bf a = load_a(AOb, CH, m0, k0, lane);
    #pragma unroll
    for (int j = 0; j < 4; ++j) {
      v16bf bb = load_b(wTo, CH, n0 + 16 * j, k0, lane);
      acc[j] = wmma_bf16(a, bb, acc[j]);
    }
  }
  const int rb  = (lane < 16) ? 0 : 8;
  const int nlo = lane & 15;
  #pragma unroll
  for (int j = 0; j < 4; ++j) {
    int n = n0 + 16 * j + nlo;
    float bias = out_b[n];
    #pragma unroll
    for (int i = 0; i < 8; ++i) {
      size_t idx = (size_t)(m0 + rb + i) * CH + n;
      out[idx] = acc[j][i] + bias + hs[idx];
    }
  }
}

// ---------------------------------------------------------------------------
// launch
// ---------------------------------------------------------------------------
extern "C" void kernel_launch(void* const* d_in, const int* in_sizes, int n_in,
                              void* d_out, int out_size, void* d_ws, size_t ws_size,
                              hipStream_t stream) {
  const float* hs       = (const float*)d_in[0];
  const float* src_mask = (const float*)d_in[1];
  const float* attn_wq  = (const float*)d_in[2];
  const float* attn_wk  = (const float*)d_in[3];
  const float* attn_wv  = (const float*)d_in[4];
  const float* out_w    = (const float*)d_in[5];
  const float* out_b    = (const float*)d_in[6];
  const float* proc_wq  = (const float*)d_in[7];
  const float* proc_wk  = (const float*)d_in[8];
  const float* proc_wv  = (const float*)d_in[9];
  const float* ww       = (const float*)d_in[10];
  const float* wb       = (const float*)d_in[11];
  float* out            = (float*)d_out;

  // workspace carve-up (~76 MB total)
  char* ws = (char*)d_ws;
  const size_t SZ_HSB = (size_t)TOK * CH * 2;          // hidden bf16
  const size_t SZ_WT  = (size_t)7 * CH * CH * 2;       // 7 transposed bf16 weights
  const size_t SZ_G   = (size_t)TOK * 4;               // gate f32
  const size_t SZ_QB  = (size_t)TOK * CH * 2;          // Q bf16
  const size_t SZ_KB  = (size_t)TOK * CH * 2;          // K bf16
  const size_t SZ_VT  = (size_t)BATCH * CH * SEQ2 * 2; // V^T bf16
  size_t off = 0;
  bf16_t* hsb  = (bf16_t*)(ws + off); off += SZ_HSB;
  bf16_t* wT   = (bf16_t*)(ws + off); off += SZ_WT;
  float*  gate = (float*) (ws + off); off += SZ_G;
  bf16_t* Qb   = (bf16_t*)(ws + off); off += SZ_QB;
  bf16_t* Kb   = (bf16_t*)(ws + off); off += SZ_KB;
  bf16_t* Vt   = (bf16_t*)(ws + off); off += SZ_VT;
  bf16_t* AOb  = (bf16_t*)(ws + off);                  // attn output bf16

  // 1) hidden -> bf16
  k_cvt<<<(TOK * CH / 8 + 255) / 256, 256, 0, stream>>>(hs, hsb);
  // 2) weight transpose + bf16 (slots: aq, pq, ak, pk, av, pv, ow)
  k_trans<<<dim3(CH / 32, CH / 32, 7), dim3(32, 8), 0, stream>>>(
      attn_wq, proc_wq, attn_wk, proc_wk, attn_wv, proc_wv, out_w, wT);
  // 3) gate
  k_gate<<<TOK / 4, 128, 0, stream>>>(hs, src_mask, ww, wb, gate);
  // 4) fused QKV projections (templated: branch-free inner loops)
  dim3 gproj(CH / 64, TOK / 64);
  k_qkv<0><<<gproj, 128, 0, stream>>>(hsb, wT, gate, Qb, Kb, Vt);
  k_qkv<1><<<gproj, 128, 0, stream>>>(hsb, wT, gate, Qb, Kb, Vt);
  k_qkv<2><<<gproj, 128, 0, stream>>>(hsb, wT, gate, Qb, Kb, Vt);
  // 5) flash attention (2048 wave-jobs, 4 waves/block)
  k_attn<<<BATCH * NHEADS * (SEQ2 / 16) / 4, 128, 0, stream>>>(Qb, Kb, Vt, AOb);
  // 6) output projection + bias + residual
  k_out<<<dim3(CH / 64, TOK / 64), 128, 0, stream>>>(
      AOb, wT + (size_t)6 * CH * CH, out_b, hs, out);
}
